// SelfPatchHead_30614526886455
// MI455X (gfx1250) — compile-verified, matrix operand order checked by reading, not run on hardware
//
#include <hip/hip_runtime.h>
#include <hip/hip_bf16.h>
#include <math.h>

typedef __hip_bfloat16 bhalf;
typedef __attribute__((ext_vector_type(16))) __bf16 v16bf;
typedef __attribute__((ext_vector_type(8)))  float  v8f;

struct Pack32 { uint4 lo, hi; };   // 32 bytes == one v16bf fragment

#define C_    768
#define H_    12
#define NT_   1568
#define B_    8
#define BT_   12544      /* B_*NT_ */
#define NLOC_ 5
#define NGLO_ 1569
#define MLOC_ 62720      /* BT_*NLOC_ */
#define MGLO_ 12552      /* B_*NGLO_ */
#define FF_   3072

__device__ __forceinline__ float waveSum(float v) {
#pragma unroll
  for (int m = 16; m > 0; m >>= 1) v += __shfl_xor(v, m, 32);
  return v;
}

// ---------------------------------------------------------------------------
// Weight convert: f32 W (K x N) -> bf16 Wt (N x K)   (one-time, reused a lot)
// ---------------------------------------------------------------------------
__global__ void wt_convert_kernel(const float* __restrict__ W, bhalf* __restrict__ Wt,
                                  int K, int N) {
  int i = blockIdx.x * 256 + threadIdx.x;
  if (i < K * N) {
    int n = i / K, k = i % K;
    Wt[i] = __float2bfloat16(W[(size_t)k * N + n]);
  }
}

__global__ void broadcast_row_kernel(const float* __restrict__ src, float* __restrict__ dst,
                                     int total) {
  int i = blockIdx.x * 256 + threadIdx.x;
  if (i < total) dst[i] = src[i % C_];
}

// ---------------------------------------------------------------------------
// Row L2 norms of x (BT_ rows of 768): one wave per row
// ---------------------------------------------------------------------------
__global__ void row_norm_kernel(const float* __restrict__ x, float* __restrict__ norms) {
  int w = (blockIdx.x * blockDim.x + threadIdx.x) >> 5;
  if (w >= BT_) return;
  int lane = threadIdx.x & 31;
  const float* r = x + (size_t)w * C_;
  float s = 0.f;
#pragma unroll
  for (int j = 0; j < 24; j++) { float t = r[lane + j * 32]; s += t * t; }
  s = waveSum(s);
  if (lane == 0) norms[w] = sqrtf(s);
}

// ---------------------------------------------------------------------------
// Cosine-sim top-4 over 8 neighbors; stores chosen GLOBAL position index.
// One wave per (b, n). Stable-descending tie order (strict >) matches lax.top_k.
// ---------------------------------------------------------------------------
__global__ void topk_kernel(const float* __restrict__ x, const int* __restrict__ local_idx,
                            const float* __restrict__ norms, int* __restrict__ topidx) {
  int w = (blockIdx.x * blockDim.x + threadIdx.x) >> 5;
  if (w >= BT_) return;
  int lane = threadIdx.x & 31;
  int b = w / NT_, n = w % NT_;
  const float* xc = x + ((size_t)b * NT_ + n) * C_;
  float cv[24];
#pragma unroll
  for (int j = 0; j < 24; j++) cv[j] = xc[lane + j * 32];
  float cn = fmaxf(norms[b * NT_ + n], 1e-12f);
  float sim[8];
#pragma unroll
  for (int s = 0; s < 8; s++) {
    int g = local_idx[n * 8 + s];
    const float* xs = x + ((size_t)b * NT_ + g) * C_;
    float d = 0.f;
#pragma unroll
    for (int j = 0; j < 24; j++) d += cv[j] * xs[lane + j * 32];
    d = waveSum(d);
    sim[s] = d / (cn * fmaxf(norms[b * NT_ + g], 1e-12f));
  }
  unsigned used = 0;
#pragma unroll
  for (int j = 0; j < 4; j++) {
    float best = -INFINITY; int bi = 0;
#pragma unroll
    for (int s = 0; s < 8; s++)
      if (!((used >> s) & 1u) && sim[s] > best) { best = sim[s]; bi = s; }
    used |= 1u << bi;
    if (lane == 0) topidx[w * 4 + j] = local_idx[n * 8 + bi];
  }
}

// ---------------------------------------------------------------------------
// LayerNorm of u = concat(cls, tokens) rows -> bf16.  MODE 0: loc, 1: glo.
// One wave per row.
// ---------------------------------------------------------------------------
template <int MODE>
__global__ void ln_u_kernel(const float* __restrict__ cls, const float* __restrict__ x,
                            const int* __restrict__ topidx,
                            const float* __restrict__ gamma, const float* __restrict__ beta,
                            bhalf* __restrict__ out, int Bb, int N) {
  int w = (blockIdx.x * blockDim.x + threadIdx.x) >> 5;
  if (w >= Bb * N) return;
  int lane = threadIdx.x & 31;
  int b = w / N, n = w % N;
  const float* src;
  if (n == 0) {
    src = cls + (size_t)b * C_;
  } else if (MODE == 0) {
    int batch = b / NT_;
    int g = topidx[b * 4 + (n - 1)];
    src = x + ((size_t)batch * NT_ + g) * C_;
  } else {
    src = x + ((size_t)b * NT_ + (n - 1)) * C_;
  }
  float v[24]; float s1 = 0.f, s2 = 0.f;
#pragma unroll
  for (int j = 0; j < 24; j++) { float t = src[lane + j * 32]; v[j] = t; s1 += t; s2 += t * t; }
  s1 = waveSum(s1); s2 = waveSum(s2);
  float m = s1 * (1.0f / C_);
  float var = s2 * (1.0f / C_) - m * m;
  float rstd = rsqrtf(var + 1e-6f);
  bhalf* o = out + (size_t)w * C_;
#pragma unroll
  for (int j = 0; j < 24; j++) {
    int c = lane + j * 32;
    o[c] = __float2bfloat16((v[j] - m) * rstd * gamma[c] + beta[c]);
  }
}

// Plain row LayerNorm.  OUTBF 1 -> bf16, 0 -> f32.
template <int OUTBF>
__global__ void ln_rows_kernel(const float* __restrict__ in, const float* __restrict__ gamma,
                               const float* __restrict__ beta, void* __restrict__ outv, int rows) {
  int w = (blockIdx.x * blockDim.x + threadIdx.x) >> 5;
  if (w >= rows) return;
  int lane = threadIdx.x & 31;
  const float* src = in + (size_t)w * C_;
  float v[24]; float s1 = 0.f, s2 = 0.f;
#pragma unroll
  for (int j = 0; j < 24; j++) { float t = src[lane + j * 32]; v[j] = t; s1 += t; s2 += t * t; }
  s1 = waveSum(s1); s2 = waveSum(s2);
  float m = s1 * (1.0f / C_);
  float rstd = rsqrtf(s2 * (1.0f / C_) - m * m + 1e-6f);
#pragma unroll
  for (int j = 0; j < 24; j++) {
    int c = lane + j * 32;
    float y = (v[j] - m) * rstd * gamma[c] + beta[c];
    if (OUTBF) ((bhalf*)outv)[(size_t)w * C_ + c] = __float2bfloat16(y);
    else       ((float*)outv)[(size_t)w * C_ + c] = y;
  }
}

// Final: LN(concat(gloCls, locCls)) -> f32 d_out, rows = MGLO_
__global__ void final_ln_kernel(const float* __restrict__ gloCls, const float* __restrict__ locCls,
                                const float* __restrict__ gamma, const float* __restrict__ beta,
                                float* __restrict__ out) {
  int w = (blockIdx.x * blockDim.x + threadIdx.x) >> 5;
  if (w >= MGLO_) return;
  int lane = threadIdx.x & 31;
  int b = w / NGLO_, n = w % NGLO_;
  const float* src = (n == 0) ? gloCls + (size_t)b * C_
                              : locCls + ((size_t)b * NT_ + (n - 1)) * C_;
  float v[24]; float s1 = 0.f, s2 = 0.f;
#pragma unroll
  for (int j = 0; j < 24; j++) { float t = src[lane + j * 32]; v[j] = t; s1 += t; s2 += t * t; }
  s1 = waveSum(s1); s2 = waveSum(s2);
  float m = s1 * (1.0f / C_);
  float rstd = rsqrtf(s2 * (1.0f / C_) - m * m + 1e-6f);
#pragma unroll
  for (int j = 0; j < 24; j++) {
    int c = lane + j * 32;
    out[(size_t)w * C_ + c] = (v[j] - m) * rstd * gamma[c] + beta[c];
  }
}

// ---------------------------------------------------------------------------
// Tiled bf16 WMMA GEMM:  Y(M,N) = X(M,K) @ Wt(N,K)^T + bias, epilogue EPI:
//   0: store f32    1: store bf16    2: tanh-GELU -> bf16
//   3: out_f32 = resid + gvec[col] * (acc + bias)
// Block = 256 threads (8 waves), tile 128x128, K-step 32.
// Wave (wm 0..3, wn 0..1) owns 32x64 = 2x4 wmma 16x16 tiles.
// ---------------------------------------------------------------------------
template <int EPI>
__global__ __launch_bounds__(256)
void gemm_bf16_kernel(const bhalf* __restrict__ X, size_t xStride,
                      const bhalf* __restrict__ Wt, const float* __restrict__ bias,
                      int M, int N, int K, void* __restrict__ outv,
                      const float* __restrict__ resid, const float* __restrict__ gvec) {
  constexpr int BM = 128, BN = 128, BK = 32, LDT = BK + 8;
  __shared__ __align__(16) bhalf sA[BM * LDT];
  __shared__ __align__(16) bhalf sB[BN * LDT];

  const int tid  = threadIdx.x;
  const int m0   = blockIdx.y * BM;
  const int n0   = blockIdx.x * BN;
  const int lane = tid & 31;
  const int wv   = tid >> 5;
  const int wm   = wv >> 1, wn = wv & 1;
  const int lm   = lane & 15, lh = lane >> 4;
  const int kb   = lh * 8;

  v8f acc[2][4];
#pragma unroll
  for (int mi = 0; mi < 2; mi++)
#pragma unroll
    for (int ni = 0; ni < 4; ni++)
      acc[mi][ni] = (v8f){0.f, 0.f, 0.f, 0.f, 0.f, 0.f, 0.f, 0.f};

  const int arow  = tid >> 1;          // 0..127
  const int ahalf = (tid & 1) * 16;    // 0 or 16 elements

  for (int k0 = 0; k0 < K; k0 += BK) {
    uint4 a0 = {0, 0, 0, 0}, a1 = {0, 0, 0, 0};
    if (m0 + arow < M) {
      const uint4* s = reinterpret_cast<const uint4*>(X + (size_t)(m0 + arow) * xStride + k0 + ahalf);
      a0 = s[0]; a1 = s[1];
    }
    uint4 b0 = {0, 0, 0, 0}, b1 = {0, 0, 0, 0};
    if (n0 + arow < N) {
      const uint4* s = reinterpret_cast<const uint4*>(Wt + (size_t)(n0 + arow) * K + k0 + ahalf);
      b0 = s[0]; b1 = s[1];
    }
    __syncthreads();
    *reinterpret_cast<uint4*>(sA + arow * LDT + ahalf)     = a0;
    *reinterpret_cast<uint4*>(sA + arow * LDT + ahalf + 8) = a1;
    *reinterpret_cast<uint4*>(sB + arow * LDT + ahalf)     = b0;
    *reinterpret_cast<uint4*>(sB + arow * LDT + ahalf + 8) = b1;
    __syncthreads();

    // Fragment loads follow the ISA 16-bit A/B layouts:
    // lane (lm,lh): elems 0..7 = K kb..kb+7, elems 8..15 = K kb+16..kb+23
    v16bf af[2], bfr[4];
#pragma unroll
    for (int mi = 0; mi < 2; mi++) {
      const bhalf* p = sA + (wm * 32 + mi * 16 + lm) * LDT + kb;
      Pack32 pk; pk.lo = *reinterpret_cast<const uint4*>(p);
                 pk.hi = *reinterpret_cast<const uint4*>(p + 16);
      af[mi] = __builtin_bit_cast(v16bf, pk);
    }
#pragma unroll
    for (int ni = 0; ni < 4; ni++) {
      const bhalf* p = sB + (wn * 64 + ni * 16 + lm) * LDT + kb;
      Pack32 pk; pk.lo = *reinterpret_cast<const uint4*>(p);
                 pk.hi = *reinterpret_cast<const uint4*>(p + 16);
      bfr[ni] = __builtin_bit_cast(v16bf, pk);
    }
#pragma unroll
    for (int mi = 0; mi < 2; mi++)
#pragma unroll
      for (int ni = 0; ni < 4; ni++)
        acc[mi][ni] = __builtin_amdgcn_wmma_f32_16x16x32_bf16(
            false, af[mi], false, bfr[ni], (short)0, acc[mi][ni], false, false);
  }

  // Epilogue.  D layout: element v, lane (lm,lh) -> row v + 8*lh, col lm.
#pragma unroll
  for (int mi = 0; mi < 2; mi++) {
#pragma unroll
    for (int ni = 0; ni < 4; ni++) {
      const int gn  = n0 + wn * 64 + ni * 16 + lm;
      const int gmb = m0 + wm * 32 + mi * 16 + lh * 8;
      const float bb = bias[gn];
#pragma unroll
      for (int v = 0; v < 8; v++) {
        const int gm = gmb + v;
        if (gm < M) {
          const size_t idx = (size_t)gm * N + gn;
          float y = acc[mi][ni][v] + bb;
          if (EPI == 0) {
            ((float*)outv)[idx] = y;
          } else if (EPI == 1) {
            ((bhalf*)outv)[idx] = __float2bfloat16(y);
          } else if (EPI == 2) {
            float t = 0.7978845608028654f * (y + 0.044715f * y * y * y);
            float g = 0.5f * y * (1.0f + tanhf(t));
            ((bhalf*)outv)[idx] = __float2bfloat16(g);
          } else {
            ((float*)outv)[idx] = resid[idx] + gvec[gn] * y;
          }
        }
      }
    }
  }
}

// ---------------------------------------------------------------------------
// Local attention: one wave per (b, h); N=5 keys, d=64.  o = softmax(qK/8) V
// ---------------------------------------------------------------------------
__global__ void attn_loc_kernel(const float* __restrict__ q, const bhalf* __restrict__ K,
                                const bhalf* __restrict__ V, bhalf* __restrict__ o) {
  int w = (blockIdx.x * blockDim.x + threadIdx.x) >> 5;
  if (w >= BT_ * H_) return;
  int lane = threadIdx.x & 31;
  int b = w / H_, h = w % H_;
  int c0 = lane * 2;
  const float* qp = q + (size_t)b * C_ + h * 64;
  float q0 = qp[c0], q1 = qp[c0 + 1];
  float sc[NLOC_];
#pragma unroll
  for (int n = 0; n < NLOC_; n++) {
    const bhalf* kp = K + ((size_t)(b * NLOC_ + n)) * C_ + h * 64;
    float d = q0 * __bfloat162float(kp[c0]) + q1 * __bfloat162float(kp[c0 + 1]);
    d = waveSum(d);
    sc[n] = d * 0.125f;
  }
  float mx = sc[0];
#pragma unroll
  for (int n = 1; n < NLOC_; n++) mx = fmaxf(mx, sc[n]);
  float s = 0.f;
#pragma unroll
  for (int n = 0; n < NLOC_; n++) { sc[n] = __expf(sc[n] - mx); s += sc[n]; }
  float inv = 1.0f / s;
  float o0 = 0.f, o1 = 0.f;
#pragma unroll
  for (int n = 0; n < NLOC_; n++) {
    const bhalf* vp = V + ((size_t)(b * NLOC_ + n)) * C_ + h * 64;
    o0 += sc[n] * __bfloat162float(vp[c0]);
    o1 += sc[n] * __bfloat162float(vp[c0 + 1]);
  }
  bhalf* op = o + (size_t)b * C_ + h * 64;
  op[c0]     = __float2bfloat16(o0 * inv);
  op[c0 + 1] = __float2bfloat16(o1 * inv);
}

// ---------------------------------------------------------------------------
// Global attention: one block per (b, h); N=1569 keys, d=64.
// ---------------------------------------------------------------------------
__global__ void attn_glo_kernel(const float* __restrict__ q, const bhalf* __restrict__ K,
                                const bhalf* __restrict__ V, bhalf* __restrict__ o) {
  int bh = blockIdx.x;
  int b = bh / H_, h = bh % H_;
  __shared__ float qs[64];
  __shared__ float sc[NGLO_];
  __shared__ float red[256];
  int t = threadIdx.x;
  if (t < 64) qs[t] = q[(size_t)b * C_ + h * 64 + t];
  __syncthreads();
  float lmax = -INFINITY;
  for (int n = t; n < NGLO_; n += 256) {
    const bhalf* kp = K + ((size_t)(b * NGLO_ + n)) * C_ + h * 64;
    float d = 0.f;
#pragma unroll
    for (int i = 0; i < 64; i++) d += qs[i] * __bfloat162float(kp[i]);
    d *= 0.125f;
    sc[n] = d;
    lmax = fmaxf(lmax, d);
  }
  red[t] = lmax; __syncthreads();
  for (int s = 128; s > 0; s >>= 1) { if (t < s) red[t] = fmaxf(red[t], red[t + s]); __syncthreads(); }
  float mx = red[0]; __syncthreads();
  float lsum = 0.f;
  for (int n = t; n < NGLO_; n += 256) { float e = __expf(sc[n] - mx); sc[n] = e; lsum += e; }
  red[t] = lsum; __syncthreads();
  for (int s = 128; s > 0; s >>= 1) { if (t < s) red[t] += red[t + s]; __syncthreads(); }
  float S = red[0]; __syncthreads();
  int c = t & 63, part = t >> 6;
  float acc = 0.f;
  for (int n = part; n < NGLO_; n += 4)
    acc += sc[n] * __bfloat162float(V[((size_t)(b * NGLO_ + n)) * C_ + h * 64 + c]);
  red[t] = acc; __syncthreads();
  if (t < 64) {
    float ov = (red[t] + red[t + 64] + red[t + 128] + red[t + 192]) / S;
    o[(size_t)b * C_ + h * 64 + t] = __float2bfloat16(ov);
  }
}

// ---------------------------------------------------------------------------
extern "C" void kernel_launch(void* const* d_in, const int* in_sizes, int n_in,
                              void* d_out, int out_size, void* d_ws, size_t ws_size,
                              hipStream_t stream) {
  (void)in_sizes; (void)n_in; (void)out_size; (void)ws_size;
  // Input order (recursive insertion order of setup_inputs):
  // 0:x 1:local_idx 2:cls_token 3..20:blk0 21..38:blk1 39:norm_g 40:norm_b
  const float* x         = (const float*)d_in[0];
  const int*   local_idx = (const int*)d_in[1];
  const float* cls_token = (const float*)d_in[2];
  const float* norm_g    = (const float*)d_in[39];
  const float* norm_b    = (const float*)d_in[40];
  const int blkBase[2] = {3, 21};
  // blk offsets: n1g 0, n1b 1, Wq 2, bq 3, Wk 4, bk 5, Wv 6, bv 7, Wp 8, bp 9,
  //              g1 10, g2 11, n2g 12, n2b 13, W1 14, b1 15, W2 16, b2 17

  char* ws = (char*)d_ws;
  size_t off = 0;
  auto alloc = [&](size_t bytes) -> char* {
    char* p = ws + off;
    off += (bytes + 255) & ~(size_t)255;
    return p;
  };

  bhalf *wQ[2], *wK[2], *wV[2], *wP[2], *w1[2], *w2[2];
  for (int blk = 0; blk < 2; blk++) {
    wQ[blk] = (bhalf*)alloc((size_t)C_ * C_ * 2);
    wK[blk] = (bhalf*)alloc((size_t)C_ * C_ * 2);
    wV[blk] = (bhalf*)alloc((size_t)C_ * C_ * 2);
    wP[blk] = (bhalf*)alloc((size_t)C_ * C_ * 2);
    w1[blk] = (bhalf*)alloc((size_t)C_ * FF_ * 2);
    w2[blk] = (bhalf*)alloc((size_t)FF_ * C_ * 2);
  }
  float* norms  = (float*)alloc((size_t)BT_ * 4);
  int*   topidx = (int*)alloc((size_t)BT_ * 4 * 4);
  float* locCls = (float*)alloc((size_t)BT_ * C_ * 4);
  float* gloCls = (float*)alloc((size_t)B_ * C_ * 4);
  bhalf* un = (bhalf*)alloc((size_t)MLOC_ * C_ * 2);
  bhalf* Kb = (bhalf*)alloc((size_t)MLOC_ * C_ * 2);
  bhalf* Vb = (bhalf*)alloc((size_t)MLOC_ * C_ * 2);
  float* qb = (float*)alloc((size_t)BT_ * C_ * 4);
  bhalf* ob = (bhalf*)alloc((size_t)BT_ * C_ * 2);
  bhalf* tb = (bhalf*)alloc((size_t)BT_ * C_ * 2);
  bhalf* h1 = (bhalf*)alloc((size_t)BT_ * FF_ * 2);

  // --- one-time weight conversion (f32 KxN -> bf16 NxK) ---
  for (int blk = 0; blk < 2; blk++) {
    int bs = blkBase[blk];
    auto conv = [&](int idx, bhalf* dst, int Kd, int Nd) {
      int total = Kd * Nd;
      wt_convert_kernel<<<(total + 255) / 256, 256, 0, stream>>>(
          (const float*)d_in[idx], dst, Kd, Nd);
    };
    conv(bs + 2,  wQ[blk], C_, C_);
    conv(bs + 4,  wK[blk], C_, C_);
    conv(bs + 6,  wV[blk], C_, C_);
    conv(bs + 8,  wP[blk], C_, C_);
    conv(bs + 14, w1[blk], C_, FF_);
    conv(bs + 16, w2[blk], FF_, C_);
  }

  broadcast_row_kernel<<<(BT_ * C_ + 255) / 256, 256, 0, stream>>>(cls_token, locCls, BT_ * C_);
  broadcast_row_kernel<<<(B_ * C_ + 255) / 256, 256, 0, stream>>>(cls_token, gloCls, B_ * C_);
  row_norm_kernel<<<(BT_ * 32 + 255) / 256, 256, 0, stream>>>(x, norms);
  topk_kernel<<<(BT_ * 32 + 255) / 256, 256, 0, stream>>>(x, local_idx, norms, topidx);

  for (int blk = 0; blk < 2; blk++) {
    int bs = blkBase[blk];
    const float* n1g = (const float*)d_in[bs + 0];
    const float* n1b = (const float*)d_in[bs + 1];
    const float* bq  = (const float*)d_in[bs + 3];
    const float* bk  = (const float*)d_in[bs + 5];
    const float* bv  = (const float*)d_in[bs + 7];
    const float* bp  = (const float*)d_in[bs + 9];
    const float* g1  = (const float*)d_in[bs + 10];
    const float* g2  = (const float*)d_in[bs + 11];
    const float* n2g = (const float*)d_in[bs + 12];
    const float* n2b = (const float*)d_in[bs + 13];
    const float* b1  = (const float*)d_in[bs + 15];
    const float* b2  = (const float*)d_in[bs + 17];

    // ----- local path: Bb = BT_, N = 5 -----
    ln_u_kernel<0><<<(MLOC_ * 32 + 255) / 256, 256, 0, stream>>>(
        locCls, x, topidx, n1g, n1b, un, BT_, NLOC_);
    dim3 gKV(C_ / 128, (MLOC_ + 127) / 128);
    gemm_bf16_kernel<1><<<gKV, 256, 0, stream>>>(un, (size_t)C_, wK[blk], bk,
                                                 MLOC_, C_, C_, Kb, nullptr, nullptr);
    gemm_bf16_kernel<1><<<gKV, 256, 0, stream>>>(un, (size_t)C_, wV[blk], bv,
                                                 MLOC_, C_, C_, Vb, nullptr, nullptr);
    dim3 gQ(C_ / 128, (BT_ + 127) / 128);
    gemm_bf16_kernel<0><<<gQ, 256, 0, stream>>>(un, (size_t)(NLOC_ * C_), wQ[blk], bq,
                                                BT_, C_, C_, qb, nullptr, nullptr);
    attn_loc_kernel<<<(BT_ * H_) / 8, 256, 0, stream>>>(qb, Kb, Vb, ob);
    gemm_bf16_kernel<3><<<gQ, 256, 0, stream>>>(ob, (size_t)C_, wP[blk], bp,
                                                BT_, C_, C_, locCls, locCls, g1);
    ln_rows_kernel<1><<<(BT_ * 32 + 255) / 256, 256, 0, stream>>>(locCls, n2g, n2b, tb, BT_);
    dim3 gF1(FF_ / 128, (BT_ + 127) / 128);
    gemm_bf16_kernel<2><<<gF1, 256, 0, stream>>>(tb, (size_t)C_, w1[blk], b1,
                                                 BT_, FF_, C_, h1, nullptr, nullptr);
    gemm_bf16_kernel<3><<<gQ, 256, 0, stream>>>(h1, (size_t)FF_, w2[blk], b2,
                                                BT_, C_, FF_, locCls, locCls, g2);

    // ----- global path: Bb = 8, N = 1569 -----
    ln_u_kernel<1><<<(MGLO_ * 32 + 255) / 256, 256, 0, stream>>>(
        gloCls, x, nullptr, n1g, n1b, un, B_, NGLO_);
    dim3 gKVg(C_ / 128, (MGLO_ + 127) / 128);
    gemm_bf16_kernel<1><<<gKVg, 256, 0, stream>>>(un, (size_t)C_, wK[blk], bk,
                                                  MGLO_, C_, C_, Kb, nullptr, nullptr);
    gemm_bf16_kernel<1><<<gKVg, 256, 0, stream>>>(un, (size_t)C_, wV[blk], bv,
                                                  MGLO_, C_, C_, Vb, nullptr, nullptr);
    dim3 gQg(C_ / 128, 1);
    gemm_bf16_kernel<0><<<gQg, 256, 0, stream>>>(un, (size_t)NGLO_ * C_, wQ[blk], bq,
                                                 B_, C_, C_, qb, nullptr, nullptr);
    attn_glo_kernel<<<B_ * H_, 256, 0, stream>>>(qb, Kb, Vb, ob);
    gemm_bf16_kernel<3><<<gQg, 256, 0, stream>>>(ob, (size_t)C_, wP[blk], bp,
                                                 B_, C_, C_, gloCls, gloCls, g1);
    ln_rows_kernel<1><<<1, 256, 0, stream>>>(gloCls, n2g, n2b, tb, B_);
    dim3 gF1g(FF_ / 128, 1);
    gemm_bf16_kernel<2><<<gF1g, 256, 0, stream>>>(tb, (size_t)C_, w1[blk], b1,
                                                  B_, FF_, C_, h1, nullptr, nullptr);
    gemm_bf16_kernel<3><<<gQg, 256, 0, stream>>>(h1, (size_t)FF_, w2[blk], b2,
                                                 B_, C_, FF_, gloCls, gloCls, g2);
  }

  final_ln_kernel<<<(MGLO_ * 32 + 255) / 256, 256, 0, stream>>>(
      gloCls, locCls, norm_g, norm_b, (float*)d_out);
}
